// BNeckConvLSTM_12687333392819
// MI455X (gfx1250) — matrix operand order repaired and checked by reading, not compiled, not run on hardware
//
#include <hip/hip_runtime.h>
#include <hip/hip_bf16.h>

// ---------------------------------------------------------------------------
// BNeckConvLSTM on MI455X (gfx1250): 3x [ConvLSTM2D(F=256) + BatchNorm]
// B=4, T=8, H=W=16, C=F=256.  Compute-bound: conv-as-implicit-GEMM with
// V_WMMA_F32_16X16X32_BF16 (bf16 inputs, fp32 accumulate).
// A-tile staging uses GLOBAL_LOAD_ASYNC_TO_LDS_B128 (ASYNCcnt) — the CDNA5
// async copy path — with s_wait_asynccnt before the double-buffer barrier.
// Recurrent GEMMs (critical path, M=1024) use split-K=4 for occupancy.
// ---------------------------------------------------------------------------

typedef __bf16 bf16;
typedef bf16  bf16x4  __attribute__((ext_vector_type(4)));
typedef bf16  bf16x8  __attribute__((ext_vector_type(8)));
typedef bf16  v16bf   __attribute__((ext_vector_type(16)));
typedef float v8f     __attribute__((ext_vector_type(8)));

#define TM 128
#define TN 128
#define TK 32

// CDNA5 async global->LDS copy of 16 bytes per lane (per-lane EXEC honored).
// LDS operand: flat shared-pointer low 32 bits == LDS byte offset (ISA 10.2).
__device__ __forceinline__ void async_cp16(const bf16* g, const bf16* l) {
    asm volatile("global_load_async_to_lds_b128 %0, %1, off"
                 :: "v"((unsigned)(size_t)l), "v"(g)
                 : "memory");
}
__device__ __forceinline__ void wait_async0() {
    asm volatile("s_wait_asynccnt 0x0" ::: "memory");
}

// ---------------------------------------------------------------------------
// fp32 -> bf16 bulk convert (vectorized x4)
// ---------------------------------------------------------------------------
__global__ __launch_bounds__(256)
void k_f32_to_bf16(const float* __restrict__ in, bf16* __restrict__ out, int n4) {
    int i = blockIdx.x * blockDim.x + threadIdx.x;
    if (i < n4) {
        float4 v = ((const float4*)in)[i];
        bf16x4 o = { (bf16)v.x, (bf16)v.y, (bf16)v.z, (bf16)v.w };
        ((bf16x4*)out)[i] = o;
    }
}

__global__ __launch_bounds__(256)
void k_clear_u32(unsigned int* __restrict__ p, int n) {
    int i = blockIdx.x * blockDim.x + threadIdx.x;
    if (i < n) p[i] = 0u;
}

// ---------------------------------------------------------------------------
// Implicit-GEMM 3x3 SAME conv, NHWC, bf16 in / fp32 out via WMMA bf16.
//   A: activations (Nimg, 16, 16, Cin) bf16        (M = Nimg*256 rows)
//   W: weights (3, 3, Cin, Nout) bf16  (K = 9*Cin, contiguous per (kh,kw))
//   out: slice-local (M, Ntot) fp32 at out + blockIdx.z*M*Ntot.
// Workgroup: 256 threads = 8 waves; tile 128x128, TK=32.
// Each wave: 4 M-tiles x 2 N-tiles of 16x16 -> 8 WMMA accumulators.
// Split-K: blockIdx.z selects a contiguous run of kc_per_slice TK-chunks.
// ---------------------------------------------------------------------------
__global__ __launch_bounds__(256)
void k_igemm_conv3x3_wmma(const bf16* __restrict__ A,
                          const bf16* __restrict__ Wt,
                          const float* __restrict__ bias,
                          float* __restrict__ out,
                          int HH, int WW, int Cin, int Ntot, int kc_per_slice) {
    __shared__ __align__(16) bf16 lds_A[2][TM][TK];       // 16 KB
    __shared__ __align__(16) bf16 lds_B[2][TN][TK + 8];   // 20 KB (Bt: [n][k], padded)

    const int tid  = threadIdx.x;
    const int lane = tid & 31;
    const int half = lane >> 4;      // 0..1
    const int lm   = lane & 15;      // 0..15
    const int wv   = tid >> 5;       // 0..7
    const int wave_m = wv & 1;       // 2 M-groups of 64
    const int wave_n = wv >> 1;      // 4 N-groups of 32

    const int m0 = blockIdx.x * TM;
    const int n0 = blockIdx.y * TN;
    const int HW = HH * WW;
    const int kc0 = blockIdx.z * kc_per_slice;           // Cin % 32 == 0
    const int Mtot = gridDim.x * TM;
    out += (size_t)blockIdx.z * Mtot * Ntot;

    // A-tile global coordinates: 128 rows x 32 ch, 2 threads per row.
    const int ar = tid >> 1;                 // 0..127
    const int ac = (tid & 1) * 16;           // 0 / 16
    const int gm  = m0 + ar;
    const int img = gm / HW;
    const int hw  = gm % HW;
    const int oh  = hw / WW, ow = hw % WW;

    // B-tile: 32 ci x 128 co, 8 threads per ci row.
    const int br = tid >> 3;                 // 0..31 (ci in chunk)
    const int bc = (tid & 7) * 16;           // co offset

    bf16x8 rb0, rb1;

    // async-copy (or zero-fill) the A tile for chunk kc into lds_A[dstbuf]
    auto issue_A = [&](int kc, int dstbuf) {
        const int kidx = kc * TK;
        const int s    = kidx / Cin;         // 0..8 = kh*3+kw
        const int ci0  = kidx % Cin;
        const int kh = s / 3, kw = s % 3;
        const int ih = oh + kh - 1, iw = ow + kw - 1;
        bf16* ldst = &lds_A[dstbuf][ar][ac];
        if (ih >= 0 && ih < HH && iw >= 0 && iw < WW) {
            const bf16* p = A + ((size_t)(img * HH + ih) * WW + iw) * Cin + ci0 + ac;
            async_cp16(p, ldst);
            async_cp16(p + 8, ldst + 8);
        } else {
            bf16x8 z;
#pragma unroll
            for (int j = 0; j < 8; ++j) z[j] = (bf16)0.0f;
            *(bf16x8*)ldst       = z;
            *(bf16x8*)(ldst + 8) = z;
        }
    };

    auto load_B = [&](int kc) {
        const int kidx = kc * TK;
        const int s    = kidx / Cin;
        const int ci0  = kidx % Cin;
        const bf16* q = Wt + ((size_t)s * Cin + (ci0 + br)) * Ntot + n0 + bc;
        rb0 = *(const bf16x8*)(q);
        rb1 = *(const bf16x8*)(q + 8);
        __builtin_prefetch((const void*)(q + (size_t)TK * Ntot), 0, 1);
    };

    auto store_B = [&](int dstbuf) {
        // transpose-on-store: Bt[n][k]; padded row (40 bf16) keeps banks spread
#pragma unroll
        for (int j = 0; j < 8; ++j) lds_B[dstbuf][bc + j][br]     = rb0[j];
#pragma unroll
        for (int j = 0; j < 8; ++j) lds_B[dstbuf][bc + 8 + j][br] = rb1[j];
    };

    v8f acc[4][2] = {};

    issue_A(kc0, 0);
    load_B(kc0);
    store_B(0);
    wait_async0();
    __syncthreads();

    for (int step = 0; step < kc_per_slice; ++step) {
        const int buf  = step & 1;
        const bool more = (step + 1 < kc_per_slice);
        if (more) { issue_A(kc0 + step + 1, buf ^ 1); load_B(kc0 + step + 1); }

        // A fragments: ISA 16-bit 16x32 layout -> a[0..7]=K[8h..8h+7], a[8..15]=K[16+8h..+7]
        v16bf af[4];
#pragma unroll
        for (int mt = 0; mt < 4; ++mt) {
            const bf16* pa = &lds_A[buf][wave_m * 64 + mt * 16 + lm][0];
            bf16x8 lo = *(const bf16x8*)(pa + 8 * half);
            bf16x8 hi = *(const bf16x8*)(pa + 16 + 8 * half);
            af[mt] = __builtin_shufflevector(lo, hi, 0,1,2,3,4,5,6,7,8,9,10,11,12,13,14,15);
        }
#pragma unroll
        for (int nt = 0; nt < 2; ++nt) {
            // B fragment: b[0..7]=B[8h..8h+7][n], b[8..15]=B[16+8h..+7][n] in Bt[n][.]
            const bf16* pb = &lds_B[buf][wave_n * 32 + nt * 16 + lm][8 * half];
            bf16x8 blo = *(const bf16x8*)(pb);
            bf16x8 bhi = *(const bf16x8*)(pb + 16);
            v16bf bfv = __builtin_shufflevector(blo, bhi, 0,1,2,3,4,5,6,7,8,9,10,11,12,13,14,15);
#pragma unroll
            for (int mt = 0; mt < 4; ++mt) {
                acc[mt][nt] = __builtin_amdgcn_wmma_f32_16x16x32_bf16(
                    false, af[mt], false, bfv, (short)0, acc[mt][nt], false, false);
            }
        }

        if (more) store_B(buf ^ 1);
        wait_async0();
        __syncthreads();
    }

    // D layout: M = r + 8*half, N = lm
#pragma unroll
    for (int mt = 0; mt < 4; ++mt) {
#pragma unroll
        for (int nt = 0; nt < 2; ++nt) {
            const int n = n0 + wave_n * 32 + nt * 16 + lm;
            const float bv = bias ? bias[n] : 0.0f;
            const int mbase = m0 + wave_m * 64 + mt * 16 + 8 * half;
            float* po = out + (size_t)mbase * Ntot + n;
#pragma unroll
            for (int r = 0; r < 8; ++r)
                po[(size_t)r * Ntot] = acc[mt][nt][r] + bv;
        }
    }
}

// ---------------------------------------------------------------------------
// Fused ConvLSTM gate/state update for one timestep; sums split-K rg slices.
// g = xg[t] + sum_s rg_s ; i,f,cc,o ; hard_sigmoid ; c = f*c + i*tanh(cc)
// h = o*tanh(c) -> h_bf16 (next recurrent conv) and hseq fp32 (for BN).
// ---------------------------------------------------------------------------
__device__ __forceinline__ float hsig(float x) {
    return fminf(fmaxf(0.2f * x + 0.5f, 0.0f), 1.0f);
}

__global__ __launch_bounds__(256)
void k_lstm_gates(const float* __restrict__ xg, const float* __restrict__ rg,
                  float* __restrict__ c, bf16* __restrict__ h16,
                  float* __restrict__ hseq, int t, int T, int HW, int F,
                  int nsplit, size_t rstride) {
    const int idx = blockIdx.x * blockDim.x + threadIdx.x;   // over B*HW*F
    const int f   = idx % F;
    const int bhw = idx / F;
    const int b   = bhw / HW;
    const int hw  = bhw % HW;
    const size_t xrow = ((size_t)(b * T + t) * HW + hw);
    const float* xr = xg + xrow * 4 * F;

    float g0 = xr[f], g1 = xr[F + f], g2 = xr[2 * F + f], g3 = xr[3 * F + f];
#pragma unroll 4
    for (int sl = 0; sl < nsplit; ++sl) {
        const float* rr = rg + sl * rstride + (size_t)bhw * 4 * F;
        g0 += rr[f];
        g1 += rr[F + f];
        g2 += rr[2 * F + f];
        g3 += rr[3 * F + f];
    }
    const float gi = hsig(g0);
    const float gf = hsig(g1);
    const float go = hsig(g3);

    const float cv = gf * c[idx] + gi * tanhf(g2);
    c[idx] = cv;
    const float hv = go * tanhf(cv);
    h16[idx] = (bf16)hv;
    hseq[xrow * F + f] = hv;
}

// ---------------------------------------------------------------------------
// BatchNorm (training mode, reduce over all but channel) — deterministic 2-pass
// ---------------------------------------------------------------------------
__global__ __launch_bounds__(256)
void k_bn_partial(const float* __restrict__ hseq, float* __restrict__ partial,
                  int chunk, int F) {
    const int f  = threadIdx.x;          // blockDim.x == F == 256
    const int m0 = blockIdx.x * chunk;
    float s = 0.0f, s2 = 0.0f;
    for (int i = 0; i < chunk; ++i) {
        const float v = hseq[(size_t)(m0 + i) * F + f];
        s += v; s2 += v * v;
    }
    partial[(size_t)blockIdx.x * 2 * F + f]     = s;
    partial[(size_t)blockIdx.x * 2 * F + F + f] = s2;
}

__global__ __launch_bounds__(256)
void k_bn_final(const float* __restrict__ partial, float* __restrict__ stats,
                int nblk, int F, int Mtot) {
    const int f = threadIdx.x;
    float s = 0.0f, s2 = 0.0f;
    for (int b = 0; b < nblk; ++b) {
        s  += partial[(size_t)b * 2 * F + f];
        s2 += partial[(size_t)b * 2 * F + F + f];
    }
    const float mu  = s / (float)Mtot;
    const float var = s2 / (float)Mtot - mu * mu;
    stats[f]     = mu;
    stats[F + f] = var;
}

__global__ __launch_bounds__(256)
void k_bn_apply(const float* __restrict__ hseq, const float* __restrict__ stats,
                const float* __restrict__ gamma, const float* __restrict__ beta,
                bf16* __restrict__ out_bf, float* __restrict__ out_f32, int F) {
    const int idx = blockIdx.x * blockDim.x + threadIdx.x;
    const int f   = idx % F;
    const float mu  = stats[f];
    const float var = stats[F + f];
    const float y = gamma[f] * (hseq[idx] - mu) * rsqrtf(var + 1e-3f) + beta[f];
    if (out_bf)  out_bf[idx]  = (bf16)y;
    if (out_f32) out_f32[idx] = y;
}

// ---------------------------------------------------------------------------
// Host orchestration
// ---------------------------------------------------------------------------
extern "C" void kernel_launch(void* const* d_in, const int* in_sizes, int n_in,
                              void* d_out, int out_size, void* d_ws, size_t ws_size,
                              hipStream_t stream) {
    (void)in_sizes; (void)n_in; (void)out_size; (void)ws_size;

    constexpr int B = 4, T = 8, H = 16, W = 16, C = 256, F = 256;
    constexpr int N4 = 4 * F;                 // 1024
    constexpr int HW = H * W;                 // 256
    constexpr int M_IN = B * T * HW;          // 8192 (input conv rows)
    constexpr int M_R  = B * HW;              // 1024 (recurrent conv rows)
    constexpr int KC   = 9 * C / TK;          // 72 K-chunks
    constexpr int NSPLIT = 4;                 // split-K for recurrent GEMMs
    constexpr size_t NW   = (size_t)9 * C * N4;       // weight elems (2,359,296)
    constexpr size_t NX   = (size_t)M_IN * C;         // activation elems (2,097,152)
    constexpr size_t NXG  = (size_t)M_IN * N4;        // xg elems
    constexpr size_t NRG  = (size_t)M_R * N4;         // rg elems per slice
    constexpr size_t NHF  = (size_t)M_R * F;          // per-step state elems
    constexpr size_t NSEQ = (size_t)M_IN * F;         // full h sequence elems

    // carve workspace (256B aligned)
    size_t off = 0;
    auto carve = [&](size_t bytes) -> void* {
        void* p = (char*)d_ws + off;
        off += (bytes + 255) & ~(size_t)255;
        return p;
    };
    bf16*  wxb     = (bf16*)carve(NW * sizeof(bf16));
    bf16*  whb     = (bf16*)carve(NW * sizeof(bf16));
    bf16*  xbf     = (bf16*)carve(NX * sizeof(bf16));
    float* xg      = (float*)carve(NXG * sizeof(float));
    float* rg      = (float*)carve(NRG * NSPLIT * sizeof(float));
    float* cst     = (float*)carve(NHF * sizeof(float));
    bf16*  h16     = (bf16*)carve(NHF * sizeof(bf16));
    float* hseq    = (float*)carve(NSEQ * sizeof(float));
    float* partial = (float*)carve((size_t)64 * 2 * F * sizeof(float));
    float* stats   = (float*)carve((size_t)2 * F * sizeof(float));

    const dim3 blk(256);
    const dim3 g_igemm_in(M_IN / TM, N4 / TN, 1);        // 64 x 8 x 1
    const dim3 g_igemm_r (M_R  / TM, N4 / TN, NSPLIT);   // 8 x 8 x 4

    for (int l = 0; l < 3; ++l) {
        const float* Wx = (const float*)d_in[1 + 5 * l];
        const float* Wh = (const float*)d_in[2 + 5 * l];
        const float* bb = (const float*)d_in[3 + 5 * l];
        const float* gg = (const float*)d_in[4 + 5 * l];
        const float* be = (const float*)d_in[5 + 5 * l];

        // convert weights to bf16
        k_f32_to_bf16<<<(NW / 4 + 255) / 256, blk, 0, stream>>>(Wx, wxb, (int)(NW / 4));
        k_f32_to_bf16<<<(NW / 4 + 255) / 256, blk, 0, stream>>>(Wh, whb, (int)(NW / 4));
        if (l == 0) {
            const float* x0 = (const float*)d_in[0];
            k_f32_to_bf16<<<(NX / 4 + 255) / 256, blk, 0, stream>>>(x0, xbf, (int)(NX / 4));
        }
        // xg = conv(x, Wx) + b over all B*T images (single K-slice, fused bias)
        k_igemm_conv3x3_wmma<<<g_igemm_in, blk, 0, stream>>>(xbf, wxb, bb, xg,
                                                             H, W, C, N4, KC);

        // zero initial c and h
        k_clear_u32<<<(NHF + 255) / 256, blk, 0, stream>>>((unsigned int*)cst, (int)NHF);
        k_clear_u32<<<(NHF / 2 + 255) / 256, blk, 0, stream>>>((unsigned int*)h16, (int)(NHF / 2));

        for (int t = 0; t < T; ++t) {
            // rg_s = partial conv(h, Wh) per K-slice (4x occupancy on critical path)
            k_igemm_conv3x3_wmma<<<g_igemm_r, blk, 0, stream>>>(h16, whb, nullptr, rg,
                                                                H, W, F, N4, KC / NSPLIT);
            // gates + state (sums the 4 slices deterministically)
            k_lstm_gates<<<(int)(NHF / 256), blk, 0, stream>>>(xg, rg, cst, h16, hseq,
                                                               t, T, HW, F, NSPLIT, NRG);
        }

        // batchnorm over (B,T,H,W)
        k_bn_partial<<<64, blk, 0, stream>>>(hseq, partial, M_IN / 64, F);
        k_bn_final<<<1, blk, 0, stream>>>(partial, stats, 64, F, M_IN);
        if (l < 2) {
            k_bn_apply<<<(int)(NSEQ / 256), blk, 0, stream>>>(hseq, stats, gg, be, xbf, nullptr, F);
        } else {
            k_bn_apply<<<(int)(NSEQ / 256), blk, 0, stream>>>(hseq, stats, gg, be, nullptr, (float*)d_out, F);
        }
    }
}
